// MultiHeadAttention_88562225643756
// MI455X (gfx1250) — matrix operand order repaired and checked by reading, not compiled, not run on hardware
//
#include <hip/hip_runtime.h>
#include <hip/hip_bf16.h>
#include <math.h>

namespace {
constexpr int B = 4, S = 2048, D = 1024, H = 16, DK = 64;
constexpr int TK = 64;  // keys staged per LDS chunk in attention

typedef __attribute__((ext_vector_type(16))) __bf16 bf16x16;
typedef __attribute__((ext_vector_type(8)))  __bf16 bf16x8;
typedef __attribute__((ext_vector_type(8)))  float  f32x8;
} // namespace

static __device__ __forceinline__ f32x8 wmma_bf16(bf16x16 a, bf16x16 b, f32x8 c) {
  return __builtin_amdgcn_wmma_f32_16x16x32_bf16(false, a, false, b, (short)0, c,
                                                 false, false);
}

// A-operand loader: row-major [M][K], 16 rows at p0, row stride ld.
// lane%16 = M, elements 0..7 = K = kBase + 8*(lane/16)+0..7, elements 8..15 = +16.
static __device__ __forceinline__ bf16x16 load_fragA(const __bf16* __restrict__ p0,
                                                     int ld, int kBase) {
  const int lane = threadIdx.x & 31;
  const int r = lane & 15;
  const int hh = lane >> 4;
  const __bf16* p = p0 + (size_t)r * ld + kBase + hh * 8;
  union { bf16x16 v; bf16x8 h2[2]; } u;
  u.h2[0] = *(const bf16x8*)(p);
  u.h2[1] = *(const bf16x8*)(p + 16);
  return u.v;
}

// B-operand loader: K-major [K][N] (row stride ld). lane%16 = N, elements 0..7 = K = kBase + 8*(lane/16)+0..7, elements 8..15 = +16.
static __device__ __forceinline__ bf16x16 load_fragB(const __bf16* __restrict__ p0,
                                                     int ld, int kBase, int nBase) {
  const int lane = threadIdx.x & 31;
  const __bf16* p = p0 + (size_t)(kBase + (lane >> 4) * 8) * ld + nBase + (lane & 15);
  bf16x16 v;
#pragma unroll
  for (int s = 0; s < 8; ++s) {
    v[s] = p[(size_t)s * ld];
    v[8 + s] = p[(size_t)(16 + s) * ld];
  }
  return v;
}

// Async global->LDS 16B copy (per lane). Tracked by ASYNCcnt.
static __device__ __forceinline__ void async_g2l_b128(uint32_t lds_off,
                                                      const void* g) {
  asm volatile("global_load_async_to_lds_b128 %0, %1, off"
               :: "v"(lds_off), "v"((unsigned long long)(uintptr_t)g)
               : "memory");
}
static __device__ __forceinline__ void wait_async0() {
  asm volatile("s_wait_asynccnt 0" ::: "memory");
}

// ---------------- elementwise converts ----------------
__global__ void k_cvt(const float* __restrict__ in, __bf16* __restrict__ out, int n) {
  int i = blockIdx.x * 256 + threadIdx.x;
  if (i < n) out[i] = (__bf16)in[i];
}

// Wo [j][d] f32  ->  Wot [d][j] bf16 (K-major for the output projection)
__global__ void k_cvt_wot(const float* __restrict__ in, __bf16* __restrict__ out) {
  int i = blockIdx.x * 256 + threadIdx.x;
  if (i >= D * D) return;
  int j = i / D, d = i - j * D;
  out[(size_t)d * D + j] = (__bf16)in[i];
}

// ---------------- QKV projection: [S,D] @ [D,DK] per (b,h) ----------------
// Each wave: 64 rows x 64 cols register tile (16 wmmas / k-step). 8 waves/block.
__global__ __launch_bounds__(256)
void k_proj(const __bf16* __restrict__ X,   // [B*S, D] bf16
            const __bf16* __restrict__ W,   // [H, D, DK] bf16 (K-major per head)
            const float*  __restrict__ bias,// [H, DK]
            __bf16* __restrict__ out,       // [B,H,S,DK] or [B,H,DK,S] if trans
            int trans) {
  const int w = threadIdx.x >> 5;
  const int lane = threadIdx.x & 31;
  const int l = lane & 15, hh = lane >> 4;
  const int h = blockIdx.y;
  const int g0 = blockIdx.x * 512 + w * 64;  // row in flattened [B*S)
  const int b = g0 / S, s0 = g0 - b * S;

  const __bf16* Wh = W + (size_t)h * D * DK;
  f32x8 acc[4][4] = {};
  for (int k = 0; k < D; k += 32) {
    __builtin_prefetch(X + (size_t)g0 * D + k + 64, 0, 0);
    bf16x16 a[4];
#pragma unroll
    for (int mi = 0; mi < 4; ++mi)
      a[mi] = load_fragA(X + (size_t)(g0 + 16 * mi) * D, D, k);
#pragma unroll
    for (int j = 0; j < 4; ++j) {
      bf16x16 bb = load_fragB(Wh, DK, k, j * 16);
#pragma unroll
      for (int mi = 0; mi < 4; ++mi)
        acc[mi][j] = wmma_bf16(a[mi], bb, acc[mi][j]);
    }
  }
#pragma unroll
  for (int j = 0; j < 4; ++j) {
    const int n = j * 16 + l;
    const float bvs = bias[h * DK + n];
#pragma unroll
    for (int mi = 0; mi < 4; ++mi) {
#pragma unroll
      for (int v = 0; v < 8; ++v) {
        const int m = 16 * mi + hh * 8 + v;
        float val = acc[mi][j][v] + bvs;
        if (!trans)
          out[(((size_t)b * H + h) * S + (s0 + m)) * DK + n] = (__bf16)val;
        else
          out[(((size_t)b * H + h) * DK + n) * S + (s0 + m)] = (__bf16)val;
      }
    }
  }
}

// ---------------- flash attention per (b,h) with async K/V staging ----------------
__global__ __launch_bounds__(256)
void k_attn(const __bf16* __restrict__ Q,   // [B,H,S,DK]
            const __bf16* __restrict__ Kt,  // [B,H,DK,S]  (K-major for scores)
            const __bf16* __restrict__ V,   // [B,H,S,DK]  (K-major for P@V)
            const int*    __restrict__ mask,// [S,S]
            __bf16* __restrict__ Cc) {      // [B,S,D] head-concat output
  __shared__ __bf16 kbuf[2][DK][TK];  // [dk][tloc] K-major score operand
  __shared__ __bf16 vbuf[2][TK][DK];  // [tloc][dk] K-major P@V operand
  __shared__ __bf16 pbuf[8][16 * 32]; // per-wave P tile (C-layout -> A-layout)

  const int tid = threadIdx.x;
  const int w = tid >> 5;
  const int lane = tid & 31;
  const int l = lane & 15, hh = lane >> 4;
  const int bh = blockIdx.y;
  const int b = bh / H, h = bh - b * H;
  const int q0 = blockIdx.x * 128 + w * 16;

  const __bf16* Qbh = Q + (size_t)bh * S * DK;
  const __bf16* Ktb = Kt + (size_t)bh * DK * S;
  const __bf16* Vbh = V + (size_t)bh * S * DK;

  const bf16x16 qa0 = load_fragA(Qbh + (size_t)q0 * DK, DK, 0);
  const bf16x16 qa1 = load_fragA(Qbh + (size_t)q0 * DK, DK, 32);

  f32x8 o[4] = {};
  float rm[8], rl[8];
#pragma unroll
  for (int v = 0; v < 8; ++v) { rm[v] = -1e30f; rl[v] = 0.0f; }

  // cooperative async stage of one 64-key chunk (8KB K + 8KB V)
  auto stage = [&](int tc, int bufi) {
    for (int c = tid; c < (DK * TK) / 8; c += 256) {  // 512 x b128
      const int row = c >> 3, seg = c & 7;            // K: row stride S
      async_g2l_b128((uint32_t)(uintptr_t)&kbuf[bufi][row][seg * 8],
                     Ktb + (size_t)row * S + tc + seg * 8);
    }
    const __bf16* vsrc = Vbh + (size_t)tc * DK;       // V chunk is contiguous
    __bf16* vdst = (__bf16*)vbuf[bufi];
    for (int c = tid; c < (TK * DK) / 8; c += 256)
      async_g2l_b128((uint32_t)(uintptr_t)(vdst + c * 8), vsrc + c * 8);
  };

  auto ldsFragB = [&](const __bf16* base, int ld, int kBase, int nBase) -> bf16x16 {
    const __bf16* p = base + (kBase + (lane >> 4) * 8) * ld + nBase + (lane & 15);
    bf16x16 v;
#pragma unroll
    for (int s = 0; s < 8; ++s) {
      v[s] = p[s * ld];
      v[8 + s] = p[(16 + s) * ld];
    }
    return v;
  };

  stage(0, 0);
  const int NC = S / TK;  // 32 chunks
  for (int cc = 0; cc < NC; ++cc) {
    const int cur = cc & 1;
    const int tc = cc * TK;
    wait_async0();        // my copies for chunk cc done
    __syncthreads();      // everyone's copies done; prior chunk fully consumed
    if (cc + 1 < NC) stage(tc + TK, cur ^ 1);

    for (int ts = 0; ts < TK; ts += 32) {   // two 32-key sub-iterations
      const int t0 = tc + ts;
      // scores: two 16x16 tiles, K-dim = DK (from LDS)
      f32x8 sc0 = {}, sc1 = {};
      sc0 = wmma_bf16(qa0, ldsFragB(&kbuf[cur][0][0], TK, 0,  ts),      sc0);
      sc0 = wmma_bf16(qa1, ldsFragB(&kbuf[cur][0][0], TK, 32, ts),      sc0);
      sc1 = wmma_bf16(qa0, ldsFragB(&kbuf[cur][0][0], TK, 0,  ts + 16), sc1);
      sc1 = wmma_bf16(qa1, ldsFragB(&kbuf[cur][0][0], TK, 32, ts + 16), sc1);

      // scale 1/sqrt(DK) then mask (mask==1 -> -1e9, reference order)
#pragma unroll
      for (int v = 0; v < 8; ++v) {
        const int srow = q0 + hh * 8 + v;
        const int* mrow = mask + (size_t)srow * S + t0;
        float x0 = sc0[v] * 0.125f;
        float x1 = sc1[v] * 0.125f;
        if (mrow[l] == 1)      x0 = -1e9f;
        if (mrow[16 + l] == 1) x1 = -1e9f;
        sc0[v] = x0; sc1[v] = x1;
      }

      // online softmax; row m = 8*hh+v lives across the 16 lanes of this half
      float fs[8];
#pragma unroll
      for (int v = 0; v < 8; ++v) {
        float mx = fmaxf(sc0[v], sc1[v]);
        for (int off = 8; off >= 1; off >>= 1)
          mx = fmaxf(mx, __shfl_xor(mx, off, 32));
        const float mn = fmaxf(rm[v], mx);
        const float f = __expf(rm[v] - mn);
        const float p0 = __expf(sc0[v] - mn);
        const float p1 = __expf(sc1[v] - mn);
        sc0[v] = p0; sc1[v] = p1;
        float sum = p0 + p1;
        for (int off = 8; off >= 1; off >>= 1)
          sum += __shfl_xor(sum, off, 32);
        rl[v] = rl[v] * f + sum;
        rm[v] = mn;
        fs[v] = f;
      }
#pragma unroll
      for (int j = 0; j < 4; ++j)
#pragma unroll
        for (int v = 0; v < 8; ++v)
          o[j][v] *= fs[v];

      // C-layout -> A-layout for P via LDS (16x32 bf16, row-major)
#pragma unroll
      for (int v = 0; v < 8; ++v) {
        const int m = hh * 8 + v;
        pbuf[w][m * 32 + l]      = (__bf16)sc0[v];
        pbuf[w][m * 32 + 16 + l] = (__bf16)sc1[v];
      }
      asm volatile("s_wait_dscnt 0" ::: "memory");
      bf16x16 pa;
      {
        union { bf16x16 v16; bf16x8 h2[2]; } u;
        u.h2[0] = *(const bf16x8*)&pbuf[w][l * 32 + hh * 8];
        u.h2[1] = *(const bf16x8*)&pbuf[w][l * 32 + 16 + hh * 8];
        pa = u.v16;
      }

      // o += P(16x32) @ V(32x64)   (V from LDS)
#pragma unroll
      for (int j = 0; j < 4; ++j)
        o[j] = wmma_bf16(pa, ldsFragB((const __bf16*)vbuf[cur], DK, ts, j * 16),
                         o[j]);
    }
    __syncthreads();  // all waves done reading buf[cur] before it is restaged
  }

  // normalize and write head-concat output
  float inv[8];
#pragma unroll
  for (int v = 0; v < 8; ++v) inv[v] = 1.0f / rl[v];
#pragma unroll
  for (int j = 0; j < 4; ++j) {
    const int col = h * DK + j * 16 + l;
#pragma unroll
    for (int v = 0; v < 8; ++v) {
      const int row = q0 + hh * 8 + v;
      Cc[((size_t)b * S + row) * D + col] = (__bf16)(o[j][v] * inv[v]);
    }
  }
}

// ---------------- output projection: [B*S,D] @ Wot[D,D] + bo -> f32 ----------------
__global__ __launch_bounds__(256)
void k_oproj(const __bf16* __restrict__ Cc,  // [B*S, D]
             const __bf16* __restrict__ Wot, // [D, D] K-major ([d][j])
             const float*  __restrict__ bo,  // [D]
             float* __restrict__ out) {      // [B*S, D] f32
  const int w = threadIdx.x >> 5;
  const int lane = threadIdx.x & 31;
  const int l = lane & 15, hh = lane >> 4;
  const int g0 = blockIdx.x * 512 + w * 64;
  const int n0 = blockIdx.y * 64;

  f32x8 acc[4][4] = {};
  for (int k = 0; k < D; k += 32) {
    __builtin_prefetch(Cc + (size_t)g0 * D + k + 64, 0, 0);
    bf16x16 a[4];
#pragma unroll
    for (int mi = 0; mi < 4; ++mi)
      a[mi] = load_fragA(Cc + (size_t)(g0 + 16 * mi) * D, D, k);
#pragma unroll
    for (int j = 0; j < 4; ++j) {
      bf16x16 bb = load_fragB(Wot, D, k, n0 + j * 16);
#pragma unroll
      for (int mi = 0; mi < 4; ++mi)
        acc[mi][j] = wmma_bf16(a[mi], bb, acc[mi][j]);
    }
  }
#pragma unroll
  for (int j = 0; j < 4; ++j) {
    const int n = n0 + j * 16 + l;
    const float bvs = bo[n];
#pragma unroll
    for (int mi = 0; mi < 4; ++mi) {
#pragma unroll
      for (int v = 0; v < 8; ++v) {
        const int m = 16 * mi + hh * 8 + v;
        out[((size_t)(g0 + m)) * D + n] = acc[mi][j][v] + bvs;
      }
    }
  }
}

extern "C" void kernel_launch(void* const* d_in, const int* in_sizes, int n_in,
                              void* d_out, int out_size, void* d_ws, size_t ws_size,
                              hipStream_t stream) {
  (void)in_sizes; (void)n_in; (void)out_size; (void)ws_size;
  const float* ctx  = (const float*)d_in[0];
  const float* val  = (const float*)d_in[1];
  const int*   mask = (const int*)d_in[2];
  const float* Wq   = (const float*)d_in[3];
  const float* bq   = (const float*)d_in[4];
  const float* Wk   = (const float*)d_in[5];
  const float* bk   = (const float*)d_in[6];
  const float* Wv   = (const float*)d_in[7];
  const float* bv   = (const float*)d_in[8];
  const float* Wo   = (const float*)d_in[9];
  const float* bo   = (const float*)d_in[10];
  float* out = (float*)d_out;

  char* ws = (char*)d_ws;
  size_t off = 0;
  auto carve = [&](size_t bytes) -> char* {
    char* p = ws + off;
    off += (bytes + 255) & ~(size_t)255;
    return p;
  };
  const size_t nX = (size_t)B * S * D;
  const size_t nW = (size_t)H * D * DK;
  const size_t nP = (size_t)B * H * S * DK;

  __bf16* Xc  = (__bf16*)carve(nX * 2);
  __bf16* Xv  = (__bf16*)carve(nX * 2);
  __bf16* Wqb = (__bf16*)carve(nW * 2);
  __bf16* Wkb = (__bf16*)carve(nW * 2);
  __bf16* Wvb = (__bf16*)carve(nW * 2);
  __bf16* Wot = (__bf16*)carve((size_t)D * D * 2);
  __bf16* Qb  = (__bf16*)carve(nP * 2);
  __bf16* Ktb = (__bf16*)carve(nP * 2);
  __bf16* Vb  = (__bf16*)carve(nP * 2);
  __bf16* Cc  = (__bf16*)carve(nX * 2);

  // 1) converts
  k_cvt<<<(int)((nX + 255) / 256), 256, 0, stream>>>(ctx, Xc, (int)nX);
  k_cvt<<<(int)((nX + 255) / 256), 256, 0, stream>>>(val, Xv, (int)nX);
  k_cvt<<<(int)((nW + 255) / 256), 256, 0, stream>>>(Wq, Wqb, (int)nW);
  k_cvt<<<(int)((nW + 255) / 256), 256, 0, stream>>>(Wk, Wkb, (int)nW);
  k_cvt<<<(int)((nW + 255) / 256), 256, 0, stream>>>(Wv, Wvb, (int)nW);
  k_cvt_wot<<<(D * D + 255) / 256, 256, 0, stream>>>(Wo, Wot);

  // 2) QKV projections (K written transposed so it is K-major for Q@K^T)
  dim3 gp(B * S / 512, H);
  k_proj<<<gp, 256, 0, stream>>>(Xc, Wqb, bq, Qb, 0);
  k_proj<<<gp, 256, 0, stream>>>(Xc, Wkb, bk, Ktb, 1);
  k_proj<<<gp, 256, 0, stream>>>(Xv, Wvb, bv, Vb, 0);

  // 3) flash attention with double-buffered async K/V staging
  dim3 ga(S / 128, B * H);
  k_attn<<<ga, 256, 0, stream>>>(Qb, Ktb, Vb, mask, Cc);

  // 4) output projection -> f32 d_out
  dim3 go(B * S / 512, D / 64);
  k_oproj<<<go, 256, 0, stream>>>(Cc, Wot, bo, out);
}
